// BiLSTMAttention_1022202216827
// MI455X (gfx1250) — matrix operand order repaired and checked by reading, not compiled, not run on hardware
//
#include <hip/hip_runtime.h>
#include <hip/hip_bf16.h>

// ---------------------------------------------------------------------------
// BiLSTM + additive attention for MI455X (gfx1250, wave32, WMMA bf16->f32,
// TDM async staging).
// Phases:
//   K0a: pack x -> bf16 pairs (A-operand form)
//   K0b: pack W/U/attn_W -> bf16 pairs in B-fragment layout [n][K/2]
//   K1 : xW = x @ W + b, TDM double-buffered A tiles, W panel LDS-resident
//   K2 : LSTM recurrence, U panel via one TDM load; per-step WMMA + prefetch
//   K3 : scores = ctx . tanh(h @ attn_W + b), attn_W panel via TDM
//   K4 : softmax over S
//   K5 : out = sum_s attn * h
// ---------------------------------------------------------------------------

#define BB   64
#define SS   2048
#define DD   128
#define HH   128
#define G4H  512      // 4*H
#define HH2  256      // 2*H
#define OO   256

typedef __attribute__((ext_vector_type(16))) __bf16        v16bf;
typedef __attribute__((ext_vector_type(8)))  float         v8f;
typedef __attribute__((ext_vector_type(8)))  unsigned int  u32x8;
typedef __attribute__((ext_vector_type(4)))  unsigned int  u32x4;
typedef __attribute__((ext_vector_type(8)))  int           i32x8;
typedef __attribute__((ext_vector_type(4)))  int           i32x4;

// ---- bf16 helpers (RNE-ish pack of two floats into one dword) -------------
static __device__ __forceinline__ unsigned int bf16_bits(float f) {
    unsigned int u = __builtin_bit_cast(unsigned int, f);
    return (u + 0x7FFFu + ((u >> 16) & 1u)) >> 16;
}
static __device__ __forceinline__ unsigned int pack2(float lo, float hi) {
    return bf16_bits(lo) | (bf16_bits(hi) << 16);
}

// ---- TDM: 1-D contiguous global -> LDS copy (CDNA5 ISA, D# tables) -----
// group0: count=1 | lds_addr | global_addr[56:0] | type=2
// group1: data_size=8B, tensor_dim0=elems, tile_dim0=elems, tile_dim1=1
static __device__ __forceinline__ void tdm_load_lds(void* lds_ptr,
                                                    const void* gptr,
                                                    unsigned nbytes) {
    const unsigned elems = nbytes >> 3;                     // 8-byte elements
    const unsigned long long ga = (unsigned long long)(size_t)gptr;
    u32x4 g0;
    g0[0] = 1u;                                             // count=1
    g0[1] = (unsigned)(size_t)lds_ptr;                      // LDS byte offset
    g0[2] = (unsigned)ga;                                   // global[31:0]
    g0[3] = (unsigned)((ga >> 32) & 0x01FFFFFFull) | 0x80000000u; // [56:32]|type=2
    i32x8 g1;
    g1[0] = (int)(3u << 16);                                // data_size = 8B
    g1[1] = (int)((elems & 0xFFFFu) << 16);                 // tensor_dim0[15:0]
    g1[2] = (int)(((elems >> 16) & 0xFFFFu) | (1u << 16));  // td0[31:16]|td1=1
    g1[3] = (int)((elems & 0xFFFFu) << 16);                 // tile_dim0
    g1[4] = 1;                                              // tile_dim1=1
    g1[5] = (int)elems;                                     // dim0_stride lo
    g1[6] = 0;
    g1[7] = 0;
    i32x4 zz = {0, 0, 0, 0};
#if defined(__clang_major__) && (__clang_major__ >= 23)
    i32x8 z8 = {0, 0, 0, 0, 0, 0, 0, 0};
    __builtin_amdgcn_tensor_load_to_lds(g0, g1, zz, zz, z8, 0);
#else
    __builtin_amdgcn_tensor_load_to_lds(g0, g1, zz, zz, 0);
#endif
}

// ---- WMMA fragment builders (wave32 layouts, CDNA5 ISA) --------
static __device__ __forceinline__ v16bf load_a_frag(const unsigned int* row,
                                                    int hi, int kbase) {
    u32x8 u;
    const int kp0 = kbase >> 1;
#pragma unroll
    for (int p = 0; p < 8; ++p)
        u[p] = row[kp0 + hi * 8 + p];
    return __builtin_bit_cast(v16bf, u);
}
static __device__ __forceinline__ v16bf load_b_frag(const unsigned int* row,
                                                    int hi, int kbase) {
    u32x8 u;
    const int kp0 = (kbase >> 1) + hi * 8;
#pragma unroll
    for (int p = 0; p < 8; ++p) u[p] = row[kp0 + p];
    return __builtin_bit_cast(v16bf, u);
}
static __device__ __forceinline__ v8f wmma_bf16(v16bf a, v16bf b, v8f c) {
    return __builtin_amdgcn_wmma_f32_16x16x32_bf16(false, a, false, b,
                                                   (short)0, c, false, false);
}

static __device__ __forceinline__ float fast_sigmoid(float x) {
    return 1.0f / (1.0f + __expf(-x));
}
static __device__ __forceinline__ float fast_tanh(float x) {
    float e = __expf(2.0f * x);
    return 1.0f - 2.0f / (e + 1.0f);
}

// ===========================================================================
// K0a: x [B*S][128] f32 -> xbf [B*S][64] bf16-pair dwords. grid 8192 x 256.
// ===========================================================================
__launch_bounds__(256)
__global__ void k0_pack_x(const float* __restrict__ x,
                          unsigned int* __restrict__ xbf) {
    const int idx0 = blockIdx.x * 1024 + threadIdx.x;
#pragma unroll
    for (int i = 0; i < 4; ++i) {
        const int idx = idx0 + i * 256;
        const float* px = x + (size_t)idx * 2;
        xbf[idx] = pack2(px[0], px[1]);
    }
}

// ===========================================================================
// K0b: pack weights into B-fragment layout [n][K/2]. grid 5 x 256.
// ===========================================================================
__launch_bounds__(256)
__global__ void k0_pack_w(const float* __restrict__ Wf, const float* __restrict__ Wb,
                          const float* __restrict__ Uf, const float* __restrict__ Ub,
                          const float* __restrict__ attnW,
                          unsigned int* __restrict__ WpkF, unsigned int* __restrict__ WpkB,
                          unsigned int* __restrict__ UpkF, unsigned int* __restrict__ UpkB,
                          unsigned int* __restrict__ AWpk) {
    const int tid = threadIdx.x;
    const float* src; unsigned int* dst; int N, K;
    switch (blockIdx.x) {
        case 0:  src = Wf;    dst = WpkF; N = G4H; K = HH;  break;
        case 1:  src = Wb;    dst = WpkB; N = G4H; K = HH;  break;
        case 2:  src = Uf;    dst = UpkF; N = G4H; K = HH;  break;
        case 3:  src = Ub;    dst = UpkB; N = G4H; K = HH;  break;
        default: src = attnW; dst = AWpk; N = OO;  K = HH2; break;
    }
    const int KP = K >> 1;
    const int cells = N * KP;                    // 32768 in all cases
    for (int c = tid; c < cells; c += 256) {
        const int n = c % N, kp = c / N;
        dst[n * KP + kp] = pack2(src[(size_t)(2 * kp) * N + n],
                                 src[(size_t)(2 * kp + 1) * N + n]);
    }
}

// ===========================================================================
// K1: xW GEMM. grid (256, dir), block 256 (8 waves).
// W panel (512x128 bf16, 128 KB) LDS-resident via one TDM load; 8 M-tiles of
// 64 rows each, double-buffered 16 KB TDM A loads overlapped with WMMA.
// ===========================================================================
__launch_bounds__(256)
__global__ void k1_xw_gemm(const unsigned int* __restrict__ xbf,
                           const unsigned int* __restrict__ WpkF,
                           const unsigned int* __restrict__ WpkB,
                           const float* __restrict__ bf_,
                           const float* __restrict__ bb_,
                           float* __restrict__ xWf,
                           float* __restrict__ xWb) {
    extern __shared__ unsigned int smem1[];
    unsigned int* Bpk  = smem1;                 // 512*64 = 128 KB
    unsigned int* Apk0 = Bpk + 512 * 64;        // 64*64  =  16 KB
    unsigned int* Apk1 = Apk0 + 64 * 64;        //           16 KB

    const int dir = blockIdx.y;
    const unsigned int* Wpk = dir ? WpkB : WpkF;
    const float* bias       = dir ? bb_  : bf_;
    float* outp             = dir ? xWb  : xWf;

    const int tid = threadIdx.x;
    const int wave = tid >> 5, lane = tid & 31;
    const int lm = lane & 15, hi = lane >> 4;

    const int TILES = 8;
    const int mt0 = blockIdx.x * TILES;         // first M-tile (of 2048/dir)

    if (tid < 32) {                             // wave 0 drives the TDM
        tdm_load_lds(Bpk, Wpk, 512 * 64 * 4);
        tdm_load_lds(Apk0, xbf + (size_t)mt0 * 4096, 64 * 64 * 4);
    }

    float biasv[4];
#pragma unroll
    for (int j = 0; j < 4; ++j) biasv[j] = bias[(wave * 4 + j) * 16 + lm];

    for (int i = 0; i < TILES; ++i) {
        if (tid < 32) {
            if (i + 1 < TILES) {
                tdm_load_lds((i & 1) ? Apk0 : Apk1,
                             xbf + (size_t)(mt0 + i + 1) * 4096, 64 * 64 * 4);
                __builtin_amdgcn_s_wait_tensorcnt(1);   // tile i (and B) done
            } else {
                __builtin_amdgcn_s_wait_tensorcnt(0);
            }
        }
        __syncthreads();
        const unsigned int* Apk = (i & 1) ? Apk1 : Apk0;

        for (int msub = 0; msub < 4; ++msub) {
            v16bf afr[4];
#pragma unroll
            for (int kc = 0; kc < 4; ++kc)
                afr[kc] = load_a_frag(Apk + (msub * 16 + lm) * 64, hi, kc * 32);
#pragma unroll
            for (int j = 0; j < 4; ++j) {
                const int nsub = wave * 4 + j;
                v8f acc = {0.f, 0.f, 0.f, 0.f, 0.f, 0.f, 0.f, 0.f};
#pragma unroll
                for (int kc = 0; kc < 4; ++kc) {
                    v16bf b = load_b_frag(Bpk + (nsub * 16 + lm) * 64, hi, kc * 32);
                    acc = wmma_bf16(afr[kc], b, acc);
                }
                const int col = nsub * 16 + lm;
                const size_t rowg = (size_t)(mt0 + i) * 64 + msub * 16 + hi * 8;
#pragma unroll
                for (int r = 0; r < 8; ++r)
                    outp[(rowg + r) * G4H + col] = acc[r] + biasv[j];
            }
        }
        __syncthreads();
    }
}

// ===========================================================================
// K2: LSTM recurrence. grid (B/16, dir), block 256 (8 waves).
// U panel (128 KB) in LDS via one TDM load; per step: z = h@U (WMMA),
// gates in VALU; prefetch of next step's xW rows during GEMM phase.
// ===========================================================================
__launch_bounds__(256)
__global__ void k2_lstm(const unsigned int* __restrict__ UpkF,
                        const unsigned int* __restrict__ UpkB,
                        const float* __restrict__ xWf,
                        const float* __restrict__ xWb,
                        float* __restrict__ hout,          // [B*S][256] f32
                        unsigned int* __restrict__ hbf) {  // [B*S][128] bf16x2
    extern __shared__ unsigned int smem2[];
    unsigned int* Upk = smem2;                       // 512*64  = 128 KB
    unsigned int* Hpk = smem2 + 512 * 64;            // 16*64   =   4 KB
    float* Cst = (float*)(Hpk + 16 * 64);            // 16*128  =   8 KB
    float* Z   = Cst + 16 * HH;                      // 16*512  =  32 KB

    const int dir = blockIdx.y;
    const int b0  = blockIdx.x * 16;
    const float* xW = dir ? xWb : xWf;

    const int tid = threadIdx.x;
    if (tid < 32) tdm_load_lds(Upk, dir ? UpkB : UpkF, 512 * 64 * 4);
    for (int i = tid; i < 16 * 64; i += 256) Hpk[i] = 0u;
    for (int i = tid; i < 16 * HH; i += 256) Cst[i] = 0.f;
    if (tid < 32) __builtin_amdgcn_s_wait_tensorcnt(0);
    __syncthreads();

    const int wave = tid >> 5, lane = tid & 31;
    const int lm = lane & 15, hi = lane >> 4;
    const int grow = tid >> 4;            // gate-phase row 0..15
    const int ghb  = (tid & 15) * 8;      // gate-phase h base

    for (int step = 0; step < SS; ++step) {
        const int t = dir ? (SS - 1 - step) : step;

        // Prefetch next step's xW rows (2 KB/row, 16 lines x 16 threads/row).
        const int t_next = dir ? (t > 0 ? t - 1 : 0) : (t < SS - 1 ? t + 1 : t);
        __builtin_prefetch(xW + ((size_t)(b0 + grow) * SS + t_next) * G4H
                              + (tid & 15) * 32, 0, 0);

        // ---- GEMM phase: z[16][512] = h @ U ----
        v16bf afr[4];
#pragma unroll
        for (int kc = 0; kc < 4; ++kc)
            afr[kc] = load_a_frag(Hpk + lm * 64, hi, kc * 32);
#pragma unroll
        for (int j = 0; j < 4; ++j) {
            const int nsub = wave * 4 + j;
            v8f acc = {0.f, 0.f, 0.f, 0.f, 0.f, 0.f, 0.f, 0.f};
#pragma unroll
            for (int kc = 0; kc < 4; ++kc) {
                v16bf b = load_b_frag(Upk + (nsub * 16 + lm) * 64, hi, kc * 32);
                acc = wmma_bf16(afr[kc], b, acc);
            }
            const int col = nsub * 16 + lm;
            const int mb = hi * 8;
#pragma unroll
            for (int r = 0; r < 8; ++r) Z[(mb + r) * G4H + col] = acc[r];
        }
        __syncthreads();

        // ---- Gate phase: thread -> (row, h[ghb..ghb+7]) ----
        const size_t hrow = (size_t)(b0 + grow) * SS + t;
        const float* xwrow = xW + hrow * G4H;
        float hv[8];
#pragma unroll
        for (int j = 0; j < 8; ++j) {
            const int h = ghb + j;
            float zi = Z[grow * G4H + h]          + xwrow[h];
            float zf = Z[grow * G4H + HH + h]     + xwrow[HH + h];
            float zg = Z[grow * G4H + 2 * HH + h] + xwrow[2 * HH + h];
            float zo = Z[grow * G4H + 3 * HH + h] + xwrow[3 * HH + h];
            float iv = fast_sigmoid(zi);
            float fv = fast_sigmoid(zf);
            float gv = fast_tanh(zg);
            float ov = fast_sigmoid(zo);
            float cv = fv * Cst[grow * HH + h] + iv * gv;
            Cst[grow * HH + h] = cv;
            hv[j] = ov * fast_tanh(cv);
        }
        float4* hp4 = (float4*)(hout + hrow * HH2 + dir * HH + ghb);
        hp4[0] = make_float4(hv[0], hv[1], hv[2], hv[3]);
        hp4[1] = make_float4(hv[4], hv[5], hv[6], hv[7]);
        unsigned int pk[4];
#pragma unroll
        for (int q = 0; q < 4; ++q) pk[q] = pack2(hv[2 * q], hv[2 * q + 1]);
        uint4* hb4 = (uint4*)(hbf + hrow * (HH2 / 2) + dir * (HH / 2) + (ghb >> 1));
        *hb4 = make_uint4(pk[0], pk[1], pk[2], pk[3]);
#pragma unroll
        for (int q = 0; q < 4; ++q) Hpk[grow * 64 + (ghb >> 1) + q] = pk[q];
        __syncthreads();
    }
}

// ===========================================================================
// K3: attention scores. grid 1024, block 256 (8 waves x 16-row strips).
// attn_W panel (128 KB, prepacked) in LDS via one TDM load; tanh*ctx fused
// into the WMMA epilogue; butterfly shuffle reduction across column lanes.
// ===========================================================================
__launch_bounds__(256)
__global__ void k3_scores(const unsigned int* __restrict__ hbf,
                          const unsigned int* __restrict__ AWpk,
                          const float* __restrict__ attnb,
                          const float* __restrict__ ctx,
                          float* __restrict__ scores) {
    extern __shared__ unsigned int smem3[];
    unsigned int* Wpk = smem3;                     // [256][128] = 128 KB
    float* biasl = (float*)(smem3 + 256 * 128);
    float* ctxl  = biasl + OO;

    const int tid = threadIdx.x;
    if (tid < 32) tdm_load_lds(Wpk, AWpk, 256 * 128 * 4);
    biasl[tid] = attnb[tid];
    ctxl[tid]  = ctx[tid];
    if (tid < 32) __builtin_amdgcn_s_wait_tensorcnt(0);
    __syncthreads();

    const int wave = tid >> 5, lane = tid & 31;
    const int lm = lane & 15, hi = lane >> 4;
    const size_t r0 = (size_t)blockIdx.x * 128 + wave * 16;

    const unsigned int* arow = hbf + (r0 + lm) * (HH2 / 2);
    v16bf afr[8];
#pragma unroll
    for (int kc = 0; kc < 8; ++kc) afr[kc] = load_a_frag(arow, hi, kc * 32);

    float sacc[8] = {0.f, 0.f, 0.f, 0.f, 0.f, 0.f, 0.f, 0.f};
    for (int nt = 0; nt < 16; ++nt) {
        v8f acc = {0.f, 0.f, 0.f, 0.f, 0.f, 0.f, 0.f, 0.f};
#pragma unroll
        for (int kc = 0; kc < 8; ++kc) {
            v16bf b = load_b_frag(Wpk + (nt * 16 + lm) * 128, hi, kc * 32);
            acc = wmma_bf16(afr[kc], b, acc);
        }
        const int col = nt * 16 + lm;
        const float bv = biasl[col], cv = ctxl[col];
#pragma unroll
        for (int r = 0; r < 8; ++r)
            sacc[r] += fast_tanh(acc[r] + bv) * cv;
    }
#pragma unroll
    for (int m = 1; m <= 8; m <<= 1)
#pragma unroll
        for (int r = 0; r < 8; ++r)
            sacc[r] += __shfl_xor(sacc[r], m, 32);
    if (lm == 0) {
#pragma unroll
        for (int r = 0; r < 8; ++r)
            scores[r0 + hi * 8 + r] = sacc[r];
    }
}

// ===========================================================================
// K4: softmax over S per batch row. grid B, block 256.
// ===========================================================================
__launch_bounds__(256)
__global__ void k4_softmax(const float* __restrict__ scores,
                           float* __restrict__ attn) {
    __shared__ float red[256];
    const int b = blockIdx.x, t = threadIdx.x;
    float m = -3.0e38f;
    for (int s = t; s < SS; s += 256) m = fmaxf(m, scores[(size_t)b * SS + s]);
    red[t] = m; __syncthreads();
    for (int o = 128; o > 0; o >>= 1) {
        if (t < o) red[t] = fmaxf(red[t], red[t + o]);
        __syncthreads();
    }
    const float mx = red[0]; __syncthreads();
    float sum = 0.f;
    for (int s = t; s < SS; s += 256) sum += __expf(scores[(size_t)b * SS + s] - mx);
    red[t] = sum; __syncthreads();
    for (int o = 128; o > 0; o >>= 1) {
        if (t < o) red[t] += red[t + o];
        __syncthreads();
    }
    const float inv = 1.0f / red[0];
    for (int s = t; s < SS; s += 256)
        attn[(size_t)b * SS + s] = __expf(scores[(size_t)b * SS + s] - mx) * inv;
}

// ===========================================================================
// K5: out[b,o] = sum_s attn[b,s] * h[b,s,o]. grid B, block 256 (o = tid).
// ===========================================================================
__launch_bounds__(256)
__global__ void k5_pool(const float* __restrict__ hout,
                        const float* __restrict__ attn,
                        float* __restrict__ out) {
    __shared__ float aw[256];
    const int b = blockIdx.x, o = threadIdx.x;
    float acc = 0.f;
    for (int s0 = 0; s0 < SS; s0 += 256) {
        __syncthreads();
        aw[o] = attn[(size_t)b * SS + s0 + o];
        __syncthreads();
#pragma unroll 8
        for (int j = 0; j < 256; ++j)
            acc += aw[j] * hout[((size_t)b * SS + s0 + j) * HH2 + o];
    }
    out[(size_t)b * HH2 + o] = acc;
}

// ===========================================================================
// Host launcher. Workspace layout (bytes, 256-aligned):
//   xWf 0 (256MB) | xWb 256MB | hout 512MB (128MB) | hbf 640MB (64MB)
//   scores 704MB (0.5MB) | attn (0.5MB) | xbf (32MB) | 5 packed weight panels
// ===========================================================================
extern "C" void kernel_launch(void* const* d_in, const int* in_sizes, int n_in,
                              void* d_out, int out_size, void* d_ws, size_t ws_size,
                              hipStream_t stream) {
    const float* x     = (const float*)d_in[0];
    const float* Wf    = (const float*)d_in[1];
    const float* Uf    = (const float*)d_in[2];
    const float* bf_   = (const float*)d_in[3];
    const float* Wb    = (const float*)d_in[4];
    const float* Ub    = (const float*)d_in[5];
    const float* bb_   = (const float*)d_in[6];
    const float* attnW = (const float*)d_in[7];
    const float* attnb = (const float*)d_in[8];
    const float* ctx   = (const float*)d_in[9];
    float* out = (float*)d_out;

    char* base = (char*)d_ws;
    float* xWf        = (float*)(base);
    float* xWb        = (float*)(base + 268435456ull);
    float* hout       = (float*)(base + 536870912ull);
    unsigned int* hbf = (unsigned int*)(base + 671088640ull);
    float* scores     = (float*)(base + 738197504ull);
    float* attn       = (float*)(base + 738721792ull);
    unsigned int* xbf = (unsigned int*)(base + 739246080ull);   // 33.5 MB
    unsigned int* WpkF = (unsigned int*)(base + 772800512ull);  // 128 KB each
    unsigned int* WpkB = (unsigned int*)(base + 772931584ull);
    unsigned int* UpkF = (unsigned int*)(base + 773062656ull);
    unsigned int* UpkB = (unsigned int*)(base + 773193728ull);
    unsigned int* AWpk = (unsigned int*)(base + 773324800ull);

    k0_pack_x<<<dim3(8192), 256, 0, stream>>>(x, xbf);
    k0_pack_w<<<dim3(5), 256, 0, stream>>>(Wf, Wb, Uf, Ub, attnW,
                                           WpkF, WpkB, UpkF, UpkB, AWpk);

    size_t sh1 = (512 * 64 + 2 * 64 * 64) * sizeof(unsigned int);   // 160 KB
    k1_xw_gemm<<<dim3(256, 2), 256, sh1, stream>>>(xbf, WpkF, WpkB, bf_, bb_,
                                                   xWf, xWb);

    size_t sh2 = (512 * 64 + 16 * 64) * sizeof(unsigned int)
               + (16 * HH + 16 * G4H) * sizeof(float);              // 172 KB
    k2_lstm<<<dim3(4, 2), 256, sh2, stream>>>(UpkF, UpkB, xWf, xWb, hout, hbf);

    size_t sh3 = 256 * 128 * sizeof(unsigned int) + 2 * OO * sizeof(float);
    k3_scores<<<dim3(1024), 256, sh3, stream>>>(hbf, AWpk, attnb, ctx, scores);

    k4_softmax<<<dim3(BB), 256, 0, stream>>>(scores, attn);
    k5_pool<<<dim3(BB), 256, 0, stream>>>(hout, attn, out);

    (void)in_sizes; (void)n_in; (void)out_size; (void)ws_size;
}
